// BiLSTM_CRF_24051816857683
// MI455X (gfx1250) — matrix operand order repaired and checked by reading, not compiled, not run on hardware
//
#include <hip/hip_runtime.h>
#include <hip/hip_bf16.h>

// ---------------------------------------------------------------------------
// BiLSTM-CRF for MI455X (gfx1250, wave32, WMMA).
//   V=12000 E=512 H=1024 (Hd=512) L=12 B=16 S=1024, G = 4*Hd = 2048
// Pipeline:
//   k0  f32->bf16 weight conversion (W_ih_{f,b}, W_hh_{f,b})
//   k1  embedding gather -> x_bf16 [B][S][E]
//   k2  gate GEMM  xg[dir][t][b][g] = x(b,t_src) . W_ih[g,:] + bias   (WMMA bf16,
//       software-pipelined B loads, double-buffered A)
//   k3  sequential LSTM recurrence, 1 workgroup per direction (WMMA bf16),
//       4-deep pipelined W_hh streaming from L2, LDS-resident h/c state
//   k4  emissions = hcat @ W_out^T + b_out
//   k5  CRF Viterbi + partition + gold + loss (single block)
// ---------------------------------------------------------------------------

#define SEQ  1024
#define BB   16
#define EE   512
#define HD   512
#define GG   2048
#define HH   1024
#define LL   12
#define BOSL 1
#define EOSL 2

typedef __attribute__((ext_vector_type(16))) __bf16 v16bf;
typedef __attribute__((ext_vector_type(8)))  float  v8f;

union BF16x16 { uint4 q[2]; v16bf v; };

// ---------------------------------------------------------------- k0: convert
__global__ __launch_bounds__(256) void f32_to_bf16_k(const float* __restrict__ src,
                                                     __bf16* __restrict__ dst, int n) {
  int i = blockIdx.x * 256 + threadIdx.x;
  if (i < n) dst[i] = (__bf16)src[i];
}

// ---------------------------------------------------------------- k1: gather
__global__ __launch_bounds__(256) void embed_k(const float* __restrict__ emb,
                                               const int* __restrict__ x1s,
                                               __bf16* __restrict__ xbf) {
  int i  = blockIdx.x * 256 + threadIdx.x;   // over B*S*E (exact multiple)
  int bs = i >> 9;                            // (b*S + s)
  int e  = i & 511;
  xbf[i] = (__bf16)emb[(size_t)x1s[bs] * EE + e];
}

// ---------------------------------------------------------------- k2: gate GEMM
// grid (S, 2dirs), 256 threads = 8 waves; each WG: M-tile = 16 rows (=batch),
// N = 2048 split 256 per wave (16 n-tiles), K = 512 (16 k-steps of 32).
// B loads double-buffered, A double-buffered across kt pairs.
__global__ __launch_bounds__(256) void gate_gemm_k(const __bf16* __restrict__ xbf,   // [B][S][E]
                                                   const __bf16* __restrict__ Wih,   // [2][G][E]
                                                   const float* __restrict__ bf_,    // [G]
                                                   const float* __restrict__ bb_,    // [G]
                                                   float* __restrict__ xg) {         // [2][S][B][G]
  const int s   = blockIdx.x;
  const int dir = blockIdx.y;
  __shared__ __bf16 xt[BB * EE];              // 16KB A tile, reused across all K,N

  const int s_src = dir ? (SEQ - 1 - s) : s;
  for (int i = threadIdx.x; i < BB * EE; i += 256) {
    int b = i >> 9, e = i & 511;
    xt[i] = xbf[((size_t)b * SEQ + s_src) * EE + e];
  }
  __syncthreads();

  const int lane = threadIdx.x & 31;
  const int wave = threadIdx.x >> 5;
  const int nlo  = lane & 15;
  const int hi   = lane >> 4;
  const __bf16* W    = Wih + (size_t)dir * GG * EE;
  const float*  bias = dir ? bb_ : bf_;
  const int g0w = wave * 256;

  v8f C[16] = {};
  BF16x16 A2[2], B2[2];

  auto loadA = [&](int kt, BF16x16& dst) {
    const __bf16* ap = xt + nlo * EE + kt * 32 + (hi << 3);
    dst.q[0] = *(const uint4*)(ap);
    dst.q[1] = *(const uint4*)(ap + 16);
  };
  auto loadB = [&](int kt, int nt, BF16x16& dst) {
    const __bf16* bp = W + (size_t)(g0w + nt * 16 + nlo) * EE + kt * 32 + (hi << 3);
    dst.q[0] = *(const uint4*)(bp);
    dst.q[1] = *(const uint4*)(bp + 16);
  };

  loadA(0, A2[0]);
  loadB(0, 0, B2[0]);
  for (int kt2 = 0; kt2 < 16; kt2 += 2) {
    {
      const int kt = kt2;
      __builtin_prefetch(W + (size_t)(g0w + nlo) * EE + ((kt + 2) & 15) * 32, 0, 1);
#pragma unroll
      for (int nt = 0; nt < 16; ++nt) {
        if (nt == 8) loadA((kt + 1) & 15, A2[1]);
        const int nkt = (nt == 15) ? ((kt + 1) & 15) : kt;   // wraparound dummy at end
        loadB(nkt, (nt + 1) & 15, B2[(nt + 1) & 1]);
        C[nt] = __builtin_amdgcn_wmma_f32_16x16x32_bf16(false, A2[0].v, false, B2[nt & 1].v,
                                                        (short)0, C[nt], false, false);
      }
    }
    {
      const int kt = kt2 + 1;
#pragma unroll
      for (int nt = 0; nt < 16; ++nt) {
        if (nt == 8) loadA((kt + 1) & 15, A2[0]);
        const int nkt = (nt == 15) ? ((kt + 1) & 15) : kt;
        loadB(nkt, (nt + 1) & 15, B2[(nt + 1) & 1]);
        C[nt] = __builtin_amdgcn_wmma_f32_16x16x32_bf16(false, A2[1].v, false, B2[nt & 1].v,
                                                        (short)0, C[nt], false, false);
      }
    }
  }

  float* out = xg + ((size_t)dir * SEQ + s) * BB * GG;
  for (int nt = 0; nt < 16; ++nt) {
    const int g  = g0w + nt * 16 + nlo;
    const float bv = bias[g];
    for (int r = 0; r < 8; ++r) {
      const int m = hi * 8 + r;               // batch row
      out[(size_t)m * GG + g] = C[nt][r] + bv;
    }
  }
}

// ---------------------------------------------------------------- k3: recurrence
// grid = 2 (dir), 512 threads = 16 waves. Dynamic LDS:
//   gates f32 [16][2048] (128KB) | c f32 [16][512] (32KB) | h bf16 [16][512] (16KB)
// W_hh streamed from L2 with a 4-deep B-fragment pipeline (3 b128-pairs in
// flight per wave) so WMMAs don't serialize on L2 latency: this loop is the
// sequential critical path of the whole problem.
#define LSTM_LDS_BYTES (BB * GG * 4 + BB * HD * 4 + BB * HD * 2)

__global__ __launch_bounds__(512) void lstm_recur_k(const __bf16* __restrict__ WhhAll, // [2][G][Hd]
                                                    const float* __restrict__ xg,      // [2][S][B][G]
                                                    const float* __restrict__ h0,      // [2][B][Hd]
                                                    const float* __restrict__ c0,      // [2][B][Hd]
                                                    __bf16* __restrict__ hcat) {       // [B][S][H]
  const int dir = blockIdx.x;
  extern __shared__ char smem_raw[];
  float*  gates = (float*)smem_raw;                                   // [16][2048]
  float*  cst   = (float*)(smem_raw + BB * GG * 4);                   // [16][512]
  __bf16* hbuf  = (__bf16*)(smem_raw + BB * GG * 4 + BB * HD * 4);    // [16][512]

  const __bf16* Whh = WhhAll + (size_t)dir * GG * HD;
  const int tid = threadIdx.x;

  for (int i = tid; i < BB * HD; i += 512) {
    cst[i]  = c0[(size_t)dir * BB * HD + i];
    hbuf[i] = (__bf16)h0[(size_t)dir * BB * HD + i];
  }
  __syncthreads();

  const int lane = tid & 31;
  const int wave = tid >> 5;
  const int nlo  = lane & 15;
  const int hi   = lane >> 4;

  for (int t = 0; t < SEQ; ++t) {
    // ---- WMMA phase: gates(16 x 2048) += h(16 x 512) @ Whh^T --------------
    v8f C[8] = {};
    BF16x16 A2[2], B4[4];

    auto loadA = [&](int kt, BF16x16& dst) {
      const __bf16* ap = hbuf + nlo * HD + kt * 32 + (hi << 3);
      dst.q[0] = *(const uint4*)(ap);
      dst.q[1] = *(const uint4*)(ap + 16);
    };
    auto loadB = [&](int kt, int nt, BF16x16& dst) {
      const __bf16* bp = Whh + (size_t)(wave * 128 + nt * 16 + nlo) * HD + kt * 32 + (hi << 3);
      dst.q[0] = *(const uint4*)(bp);
      dst.q[1] = *(const uint4*)(bp + 16);
    };

    loadA(0, A2[0]);
    loadB(0, 0, B4[0]);
    loadB(0, 1, B4[1]);
    loadB(0, 2, B4[2]);
    for (int kt2 = 0; kt2 < 16; kt2 += 2) {
      {
        const int kt = kt2;
#pragma unroll
        for (int nt = 0; nt < 8; ++nt) {
          if (nt == 4) loadA((kt + 1) & 15, A2[1]);
          const int nn  = nt + 3;                       // prefetch distance 3
          const int nkt = (nn >= 8) ? ((kt + 1) & 15) : kt;
          loadB(nkt, nn & 7, B4[nn & 3]);
          C[nt] = __builtin_amdgcn_wmma_f32_16x16x32_bf16(false, A2[0].v, false, B4[nt & 3].v,
                                                          (short)0, C[nt], false, false);
        }
      }
      {
        const int kt = kt2 + 1;
#pragma unroll
        for (int nt = 0; nt < 8; ++nt) {
          if (nt == 4) loadA((kt + 1) & 15, A2[0]);
          const int nn  = nt + 3;
          const int nkt = (nn >= 8) ? ((kt + 1) & 15) : kt;
          loadB(nkt, nn & 7, B4[nn & 3]);
          C[nt] = __builtin_amdgcn_wmma_f32_16x16x32_bf16(false, A2[1].v, false, B4[nt & 3].v,
                                                          (short)0, C[nt], false, false);
        }
      }
    }
    for (int nt = 0; nt < 8; ++nt) {
      const int g = wave * 128 + nt * 16 + nlo;
      for (int r = 0; r < 8; ++r) {
        const int m = hi * 8 + r;
        gates[m * GG + g] = C[nt][r];
      }
    }
    __syncthreads();

    // ---- combine phase: nonlinearity + state update -----------------------
    {
      const int j = tid;                       // 0..511
      const float* xgt = xg + ((size_t)dir * SEQ + t) * BB * GG;
      const int pos = dir ? (SEQ - 1 - t) : t;
      for (int b = 0; b < BB; ++b) {
        const float* xr = xgt + (size_t)b * GG;
        const float* gr = gates + b * GG;
        float iv = gr[j]            + xr[j];
        float fv = gr[HD + j]       + xr[HD + j];
        float gv = gr[2 * HD + j]   + xr[2 * HD + j];
        float ov = gr[3 * HD + j]   + xr[3 * HD + j];
        float c  = cst[b * HD + j];
        float si = 1.0f / (1.0f + __expf(-iv));
        float sf = 1.0f / (1.0f + __expf(-fv));
        float so = 1.0f / (1.0f + __expf(-ov));
        c = sf * c + si * tanhf(gv);
        float h = so * tanhf(c);
        cst[b * HD + j]  = c;
        hbuf[b * HD + j] = (__bf16)h;
        hcat[((size_t)b * SEQ + pos) * HH + dir * HD + j] = (__bf16)h;
      }
    }
    __syncthreads();
  }
}

// ---------------------------------------------------------------- k4: emissions
__global__ __launch_bounds__(256) void emissions_k(const __bf16* __restrict__ hcat, // [B][S][H]
                                                   const float* __restrict__ Wout,  // [L][H]
                                                   const float* __restrict__ bout,  // [L]
                                                   float* __restrict__ emis) {      // [B][S][L]
  __shared__ float Ws[LL * HH];
  __shared__ float bs[LL];
  for (int i = threadIdx.x; i < LL * HH; i += 256) Ws[i] = Wout[i];
  if (threadIdx.x < LL) bs[threadIdx.x] = bout[threadIdx.x];
  __syncthreads();

  const int bsid = blockIdx.x * 256 + threadIdx.x;   // (b*S + s), exact
  const __bf16* hr = hcat + (size_t)bsid * HH;
  float acc[LL];
  for (int l = 0; l < LL; ++l) acc[l] = bs[l];
  for (int k = 0; k < HH; ++k) {
    const float hv = (float)hr[k];
    for (int l = 0; l < LL; ++l) acc[l] += hv * Ws[l * HH + k];
  }
  for (int l = 0; l < LL; ++l) emis[(size_t)bsid * LL + l] = acc[l];
}

// ---------------------------------------------------------------- k5: CRF
// single block, 256 threads: thread = b*16 + l (l < 12 active)
__global__ __launch_bounds__(256) void crf_k(const float* __restrict__ emis,  // [B][S][L]
                                             const float* __restrict__ trans, // [L][L]
                                             const float* __restrict__ mask,  // [B][S]
                                             const int* __restrict__ tags,    // [B][S]
                                             int* __restrict__ bp,            // [S-1][B][L]
                                             float* __restrict__ out) {
  __shared__ float T[LL * LL];
  __shared__ float alph[BB][LL];
  __shared__ float endsc[BB][LL];
  __shared__ float partit[BB], gold[BB];
  __shared__ int   btags[BB];

  const int tid = threadIdx.x;
  const int b = tid >> 4, l = tid & 15;
  const bool act = (l < LL);
  if (tid < LL * LL) T[tid] = trans[tid];
  __syncthreads();

  if (act) alph[b][l] = T[BOSL * LL + l] + emis[(size_t)b * SEQ * LL + l];
  __syncthreads();

  for (int t = 1; t < SEQ; ++t) {
    float nv = 0.0f;
    if (act) {
      float m = -3.4e38f; int am = 0;
      for (int lp = 0; lp < LL; ++lp) {
        const float sc = alph[b][lp] + T[lp * LL + l];
        if (sc > m) { m = sc; am = lp; }
      }
      const float e  = emis[((size_t)b * SEQ + t) * LL + l];
      const float mt = mask[b * SEQ + t];
      nv = mt * (m + e) + (1.0f - mt) * alph[b][l];
      bp[((size_t)(t - 1) * BB + b) * LL + l] = am;
    }
    __syncthreads();
    if (act) alph[b][l] = nv;
    __syncthreads();
  }

  if (act) endsc[b][l] = alph[b][l] + T[l * LL + EOSL];
  __syncthreads();

  if (l == 0) {
    float m = -3.4e38f; int am = 0;
    for (int lp = 0; lp < LL; ++lp) { const float v = endsc[b][lp]; if (v > m) { m = v; am = lp; } }
    float s = 0.0f;
    for (int lp = 0; lp < LL; ++lp) s += __expf(endsc[b][lp] - m);
    partit[b] = m + __logf(s);
    btags[b]  = am;
    out[b]    = m;                                  // max_final_scores
  }
  __syncthreads();

  if (l == 0) {
    // backtrace
    int tag = btags[b];
    out[BB + b * SEQ + (SEQ - 1)] = (float)tag;
    for (int t = SEQ - 2; t >= 0; --t) {
      tag = bp[((size_t)t * BB + b) * LL + tag];
      out[BB + b * SEQ + t] = (float)tag;
    }
    // gold score
    const int* tg = tags + b * SEQ;
    float g = emis[(size_t)b * SEQ * LL + tg[0]] + T[BOSL * LL + tg[0]];
    float msum = mask[b * SEQ];
    for (int t = 1; t < SEQ; ++t) {
      const float mt = mask[b * SEQ + t];
      g += mt * (emis[((size_t)b * SEQ + t) * LL + tg[t]] + T[tg[t - 1] * LL + tg[t]]);
      msum += mt;
    }
    const int lastidx = (int)msum - 1;
    g += T[tg[lastidx] * LL + EOSL];
    gold[b] = g;
  }
  __syncthreads();

  if (tid == 0) {
    float loss = 0.0f;
    for (int bb2 = 0; bb2 < BB; ++bb2) loss += partit[bb2] - gold[bb2];
    out[BB + BB * SEQ] = loss;                      // index 16400
  }
}

// ---------------------------------------------------------------------------
extern "C" void kernel_launch(void* const* d_in, const int* in_sizes, int n_in,
                              void* d_out, int out_size, void* d_ws, size_t ws_size,
                              hipStream_t stream) {
  const float* emb    = (const float*)d_in[0];
  const float* W_ih_f = (const float*)d_in[1];
  const float* W_hh_f = (const float*)d_in[2];
  const float* b_f    = (const float*)d_in[3];
  const float* W_ih_b = (const float*)d_in[4];
  const float* W_hh_b = (const float*)d_in[5];
  const float* b_b    = (const float*)d_in[6];
  const float* W_out  = (const float*)d_in[7];
  const float* b_out  = (const float*)d_in[8];
  const float* trans  = (const float*)d_in[9];
  const float* h0     = (const float*)d_in[10];
  const float* c0     = (const float*)d_in[11];
  const float* mask   = (const float*)d_in[12];
  const int*   x1s    = (const int*)d_in[13];
  const int*   tags   = (const int*)d_in[14];
  float* out = (float*)d_out;

  char* ws = (char*)d_ws;
  size_t off = 0;
  auto alloc = [&](size_t bytes) -> void* {
    void* p = ws + off;
    off = (off + bytes + 255) & ~(size_t)255;
    return p;
  };
  __bf16* xbf  = (__bf16*)alloc((size_t)BB * SEQ * EE * 2);       //  16 MB
  __bf16* Wih  = (__bf16*)alloc((size_t)2 * GG * EE * 2);         //   4 MB
  __bf16* Whh  = (__bf16*)alloc((size_t)2 * GG * HD * 2);         //   4 MB
  float*  xg   = (float*) alloc((size_t)2 * SEQ * BB * GG * 4);   // 256 MB
  __bf16* hcat = (__bf16*)alloc((size_t)BB * SEQ * HH * 2);       //  32 MB
  float*  emis = (float*) alloc((size_t)BB * SEQ * LL * 4);       // 768 KB
  int*    bp   = (int*)   alloc((size_t)(SEQ - 1) * BB * LL * 4); // 768 KB

  hipFuncSetAttribute(reinterpret_cast<const void*>(lstm_recur_k),
                      hipFuncAttributeMaxDynamicSharedMemorySize, LSTM_LDS_BYTES);

  const int WN = GG * EE;          // 1048576 per weight matrix
  f32_to_bf16_k<<<WN / 256, 256, 0, stream>>>(W_ih_f, Wih,          WN);
  f32_to_bf16_k<<<WN / 256, 256, 0, stream>>>(W_ih_b, Wih + WN,     WN);
  f32_to_bf16_k<<<WN / 256, 256, 0, stream>>>(W_hh_f, Whh,          WN);
  f32_to_bf16_k<<<WN / 256, 256, 0, stream>>>(W_hh_b, Whh + WN,     WN);

  embed_k<<<(BB * SEQ * EE) / 256, 256, 0, stream>>>(emb, x1s, xbf);

  gate_gemm_k<<<dim3(SEQ, 2), 256, 0, stream>>>(xbf, Wih, b_f, b_b, xg);

  lstm_recur_k<<<2, 512, LSTM_LDS_BYTES, stream>>>(Whh, xg, h0, c0, hcat);

  emissions_k<<<(BB * SEQ) / 256, 256, 0, stream>>>(hcat, W_out, b_out, emis);

  crf_k<<<1, 256, 0, stream>>>(emis, trans, mask, tags, bp, out);
}
